// MSFusion_After_Layer_67619965108900
// MI455X (gfx1250) — matrix operand order repaired and checked by reading, not compiled, not run on hardware
//
#include <hip/hip_runtime.h>

// ---------------- problem constants (match reference) ----------------
constexpr int D_     = 256;
constexpr int NH_    = 8;
constexpr int HD_    = 32;
constexpr int NP_    = 4;
constexpr int NL_    = 3;
constexpr int R_     = 8;
constexpr int SUMHW_ = 64*64 + 32*32 + 16*16;   // 5376
constexpr int LQ_    = R_ + SUMHW_;             // 5384
constexpr int B_     = 2;
constexpr int FH_    = 128;
constexpr int FW_    = 128;
constexpr int HID_   = 1024;

// ---------------- types for CDNA5 WMMA ----------------
typedef __bf16 bf16_t;
typedef __attribute__((ext_vector_type(16))) __bf16 v16bf;
typedef __attribute__((ext_vector_type(8)))  __bf16 v8bf;
typedef __attribute__((ext_vector_type(8)))  float  v8f;

__device__ __forceinline__ bf16_t f2bf(float f) {
  unsigned u = __builtin_bit_cast(unsigned, f);
  u += 0x7FFFu + ((u >> 16) & 1u);               // round-to-nearest-even
  unsigned short s = (unsigned short)(u >> 16);
  return __builtin_bit_cast(bf16_t, s);
}
__device__ __forceinline__ float bf2f(bf16_t h) {
  unsigned short s = __builtin_bit_cast(unsigned short, h);
  unsigned u = ((unsigned)s) << 16;
  return __builtin_bit_cast(float, u);
}
__device__ __forceinline__ float wave_sum32(float v) {
  #pragma unroll
  for (int o = 16; o > 0; o >>= 1) v += __shfl_xor(v, o, 32);
  return v;
}

// A operand: per-lane chunks {k+c*8 .. +7} and {k+c*8+16 .. +23}  (p already offset by c*8)
__device__ __forceinline__ v16bf load_a16(const bf16_t* p) {
  v8bf a0 = *reinterpret_cast<const v8bf*>(p);
  v8bf a1 = *reinterpret_cast<const v8bf*>(p + 16);
  v16bf a;
  #pragma unroll
  for (int i = 0; i < 8; ++i) { a[i] = a0[i]; a[8 + i] = a1[i]; }
  return a;
}
// B operand: per-lane chunks {k+c*8 .. +7} and {k+c*8+16 .. +23}  (p already offset by c*8)
__device__ __forceinline__ v16bf load_b16(const bf16_t* p) {
  return load_a16(p);
}
__device__ __forceinline__ v8bf zero8bf() {
  bf16_t z = __builtin_bit_cast(bf16_t, (unsigned short)0);
  v8bf a;
  #pragma unroll
  for (int i = 0; i < 8; ++i) a[i] = z;
  return a;
}

// ---------------- LN kernels (one wave per token, D=256 -> 8 elems/lane) ----------------
__global__ void ln_src_kernel(const float* __restrict__ src, const float* __restrict__ pos,
                              const float* __restrict__ w,
                              bf16_t* __restrict__ query_bf, bf16_t* __restrict__ valin_bf) {
  int tok = blockIdx.x * 4 + (threadIdx.x >> 5);
  if (tok >= B_ * LQ_) return;
  int lane = threadIdx.x & 31, c0 = lane * 8;
  const float* xr = src + (size_t)tok * D_ + c0;
  float v[8];
  #pragma unroll
  for (int i = 0; i < 8; ++i) v[i] = xr[i];
  float s = 0.f;
  #pragma unroll
  for (int i = 0; i < 8; ++i) s += v[i];
  float mean = wave_sum32(s) * (1.f / 256.f);
  float q = 0.f;
  #pragma unroll
  for (int i = 0; i < 8; ++i) { float d = v[i] - mean; q += d * d; }
  float rstd = rsqrtf(wave_sum32(q) * (1.f / 256.f) + 1e-6f);
  const float* wr = w + c0;
  const float* pr = pos + (size_t)tok * D_ + c0;
  int b = tok / LQ_, qi = tok % LQ_;
  float y[8];
  #pragma unroll
  for (int i = 0; i < 8; ++i) {
    y[i] = (v[i] - mean) * rstd * wr[i];
    query_bf[(size_t)tok * D_ + c0 + i] = f2bf(y[i] + pr[i]);
  }
  if (qi >= R_) {
    size_t vrow = (size_t)b * SUMHW_ + (qi - R_);
    #pragma unroll
    for (int i = 0; i < 8; ++i) valin_bf[vrow * D_ + c0 + i] = f2bf(y[i]);
  }
}

__global__ void ln_bf16_kernel(const float* __restrict__ x, const float* __restrict__ w,
                               bf16_t* __restrict__ out_bf, int ntok) {
  int tok = blockIdx.x * 4 + (threadIdx.x >> 5);
  if (tok >= ntok) return;
  int lane = threadIdx.x & 31, c0 = lane * 8;
  const float* xr = x + (size_t)tok * D_ + c0;
  float v[8];
  #pragma unroll
  for (int i = 0; i < 8; ++i) v[i] = xr[i];
  float s = 0.f;
  #pragma unroll
  for (int i = 0; i < 8; ++i) s += v[i];
  float mean = wave_sum32(s) * (1.f / 256.f);
  float q = 0.f;
  #pragma unroll
  for (int i = 0; i < 8; ++i) { float d = v[i] - mean; q += d * d; }
  float rstd = rsqrtf(wave_sum32(q) * (1.f / 256.f) + 1e-6f);
  const float* wr = w + c0;
  #pragma unroll
  for (int i = 0; i < 8; ++i)
    out_bf[(size_t)tok * D_ + c0 + i] = f2bf((v[i] - mean) * rstd * wr[i]);
}

// res = a + badd; ln_bf = LN(res)*w   (residual + pre-MLP layernorm)
__global__ void residual_ln_kernel(const float* __restrict__ a, const float* __restrict__ badd,
                                   const float* __restrict__ w, float* __restrict__ res,
                                   bf16_t* __restrict__ ln_bf, int ntok) {
  int tok = blockIdx.x * 4 + (threadIdx.x >> 5);
  if (tok >= ntok) return;
  int lane = threadIdx.x & 31, c0 = lane * 8;
  const float* ar = a + (size_t)tok * D_ + c0;
  const float* br = badd + (size_t)tok * D_ + c0;
  float v[8];
  #pragma unroll
  for (int i = 0; i < 8; ++i) v[i] = ar[i] + br[i];
  float s = 0.f;
  #pragma unroll
  for (int i = 0; i < 8; ++i) s += v[i];
  float mean = wave_sum32(s) * (1.f / 256.f);
  float q = 0.f;
  #pragma unroll
  for (int i = 0; i < 8; ++i) { float d = v[i] - mean; q += d * d; }
  float rstd = rsqrtf(wave_sum32(q) * (1.f / 256.f) + 1e-6f);
  const float* wr = w + c0;
  #pragma unroll
  for (int i = 0; i < 8; ++i) {
    res[(size_t)tok * D_ + c0 + i] = v[i];
    ln_bf[(size_t)tok * D_ + c0 + i] = f2bf((v[i] - mean) * rstd * wr[i]);
  }
}

// ---------------- casts ----------------
__global__ void cast_f2bf_kernel(const float* __restrict__ in, bf16_t* __restrict__ out, int n) {
  int i = blockIdx.x * 256 + threadIdx.x;
  if (i < n) out[i] = f2bf(in[i]);
}
// conv_w (L*9, cin, cout) f32 -> (L*9, cout, cin) bf16
__global__ void convw_transpose_kernel(const float* __restrict__ in, bf16_t* __restrict__ out, int n) {
  int o = blockIdx.x * 256 + threadIdx.x;
  if (o >= n) return;
  int lt = o >> 16, r = o & 65535, cout = r >> 8, cin = r & 255;
  out[o] = f2bf(in[((size_t)lt << 16) + (cin << 8) + cout]);
}

// ---------------- generic bf16 WMMA GEMM (K must be 256): C = A[M,256] @ W[N,256]^T + bias
// A m-tile staged once per block in LDS (shared by the block's 4 waves).
__global__ void gemm_bf16_kernel(const bf16_t* __restrict__ A, const bf16_t* __restrict__ W,
                                 const float* __restrict__ bias, float* __restrict__ C,
                                 int M, int N) {
  constexpr int K = 256;
  __shared__ alignas(16) bf16_t As[16 * K];      // 8 KB
  int tid = threadIdx.x;
  int wave = tid >> 5, lane = tid & 31;
  int mt = blockIdx.x;
  // cooperative stage: 4096 bf16 = 512 v8bf chunks, 128 threads -> 4 each
  #pragma unroll
  for (int e = tid * 8; e < 16 * K; e += 128 * 8) {
    int row = e >> 8, col = e & 255;
    *reinterpret_cast<v8bf*>(&As[e]) =
        *reinterpret_cast<const v8bf*>(A + (size_t)(mt * 16 + row) * K + col);
  }
  __syncthreads();
  int nt = blockIdx.y * 4 + wave;
  if (nt * 16 >= N) return;
  int mlo = lane & 15, ch = lane >> 4;
  const bf16_t* arow = &As[mlo * K + ch * 8];
  const bf16_t* brow = W + (size_t)(nt * 16 + mlo) * K + ch * 8;
  v8f acc = {};
  #pragma unroll
  for (int k0 = 0; k0 < K; k0 += 32) {
    v16bf a = load_a16(arow + k0);
    v16bf b = load_b16(brow + k0);
    acc = __builtin_amdgcn_wmma_f32_16x16x32_bf16(false, a, false, b, (short)0, acc, false, false);
  }
  float bs = bias ? bias[nt * 16 + mlo] : 0.f;
  int rbase = ch * 8;
  #pragma unroll
  for (int r = 0; r < 8; ++r)
    C[(size_t)(mt * 16 + rbase + r) * N + nt * 16 + mlo] = acc[r] + bs;
}

// ---------------- deformable sampling: one wave per (b,q,head), lanes = 32 channels ----------------
__global__ void deform_sample_kernel(const float* __restrict__ value, const float* __restrict__ off,
                                     const float* __restrict__ logits, const float* __restrict__ ref,
                                     float* __restrict__ ao_f, bf16_t* __restrict__ ao_bf) {
  int gid = blockIdx.x * 4 + (threadIdx.x >> 5);
  if (gid >= B_ * LQ_ * NH_) return;
  int lane = threadIdx.x & 31;
  int h = gid % NH_;
  int t = gid / NH_;
  int q = t % LQ_, b = t / LQ_;
  const int Hl[3] = {64, 32, 16}, Wl[3] = {64, 32, 16}, st[3] = {0, 4096, 5120};

  const float* lg = logits + (size_t)(b * LQ_ + q) * (NH_ * NL_ * NP_) + h * NL_ * NP_;
  float la[12], mx = -1e30f;
  #pragma unroll
  for (int i = 0; i < 12; ++i) { la[i] = lg[i]; mx = fmaxf(mx, la[i]); }
  float se = 0.f;
  #pragma unroll
  for (int i = 0; i < 12; ++i) { la[i] = __expf(la[i] - mx); se += la[i]; }
  float inv = 1.f / se;

  const float* ofp = off + (size_t)(b * LQ_ + q) * (NH_ * NL_ * NP_ * 2) + h * NL_ * NP_ * 2;
  const float* rp  = ref + (size_t)(b * LQ_ + q) * (NL_ * 2);
  float acc = 0.f;
  for (int l = 0; l < NL_; ++l) {
    int Wd = Wl[l], Hd = Hl[l];
    float rx = rp[l * 2 + 0], ry = rp[l * 2 + 1];
    size_t vbase = ((size_t)(b * SUMHW_ + st[l]) * NH_ + h) * HD_ + lane;
    #pragma unroll
    for (int p = 0; p < NP_; ++p) {
      float ox = ofp[(l * NP_ + p) * 2 + 0], oy = ofp[(l * NP_ + p) * 2 + 1];
      float x = rx * Wd + ox - 0.5f;       // (ref + off/norm) * size - 0.5
      float y = ry * Hd + oy - 0.5f;
      float xf = floorf(x), yf = floorf(y);
      float lx = x - xf, ly = y - yf;
      int x0 = (int)xf, y0 = (int)yf;
      float g00 = 0.f, g10 = 0.f, g01 = 0.f, g11 = 0.f;
      if (y0 >= 0 && y0 < Hd) {
        if (x0 >= 0 && x0 < Wd)         g00 = value[vbase + (size_t)(y0 * Wd + x0) * D_];
        if (x0 + 1 >= 0 && x0 + 1 < Wd) g10 = value[vbase + (size_t)(y0 * Wd + x0 + 1) * D_];
      }
      if (y0 + 1 >= 0 && y0 + 1 < Hd) {
        if (x0 >= 0 && x0 < Wd)         g01 = value[vbase + (size_t)((y0 + 1) * Wd + x0) * D_];
        if (x0 + 1 >= 0 && x0 + 1 < Wd) g11 = value[vbase + (size_t)((y0 + 1) * Wd + x0 + 1) * D_];
      }
      float smp = g00 * (1 - lx) * (1 - ly) + g10 * lx * (1 - ly) +
                  g01 * (1 - lx) * ly + g11 * lx * ly;
      acc += (la[l * NP_ + p] * inv) * smp;
    }
  }
  size_t oi = (size_t)(b * LQ_ + q) * D_ + h * HD_ + lane;
  ao_f[oi]  = acc;
  ao_bf[oi] = f2bf(acc);
}

// ---------------- bilinear upsample(prev, 2x) + feat add -> bf16 conv input ----------------
__global__ void upsample_add_kernel(const float* __restrict__ featF, const bf16_t* __restrict__ featB,
                                    long feat_bstride, long feat_row_off,
                                    const float* __restrict__ prev, long prev_bstride, long prev_row_off,
                                    int IH, int IW, int OH, int OW, bf16_t* __restrict__ out) {
  long idx = (long)blockIdx.x * blockDim.x + threadIdx.x;
  long total = (long)B_ * OH * OW * D_;
  if (idx >= total) return;
  int c = idx & 255;
  long t = idx >> 8;
  int x = (int)(t % OW); t /= OW;
  int y = (int)(t % OH);
  int b = (int)(t / OH);
  float fy = (y + 0.5f) * ((float)IH / OH) - 0.5f; fy = fmaxf(fy, 0.f);
  int y0 = min((int)floorf(fy), IH - 1), y1 = min(y0 + 1, IH - 1);
  float ty = fy - y0;
  float fx = (x + 0.5f) * ((float)IW / OW) - 0.5f; fx = fmaxf(fx, 0.f);
  int x0 = min((int)floorf(fx), IW - 1), x1 = min(x0 + 1, IW - 1);
  float tx = fx - x0;
  const float* pv = prev + (size_t)b * prev_bstride + (size_t)prev_row_off * D_;
  float p00 = pv[(size_t)(y0 * IW + x0) * D_ + c];
  float p01 = pv[(size_t)(y0 * IW + x1) * D_ + c];
  float p10 = pv[(size_t)(y1 * IW + x0) * D_ + c];
  float p11 = pv[(size_t)(y1 * IW + x1) * D_ + c];
  float itp = (p00 * (1 - tx) + p01 * tx) * (1 - ty) + (p10 * (1 - tx) + p11 * tx) * ty;
  size_t fidx = (size_t)b * feat_bstride + (feat_row_off + (size_t)y * OW + x) * D_ + c;
  float fv = featF ? featF[fidx] : bf2f(featB[fidx]);
  out[((size_t)(b * OH + y) * OW + x) * D_ + c] = f2bf(fv + itp);
}

// ---------------- conv3x3 SAME, implicit GEMM (K = 9*256) ----------------
// Zero-padded 3x18x256 halo activation tile staged once per block in LDS (shared by 16 waves).
__global__ void conv3x3_kernel(const bf16_t* __restrict__ in, const bf16_t* __restrict__ wt,
                               const float* __restrict__ bias, float* __restrict__ out,
                               int Hc, int Wc) {
  __shared__ alignas(16) bf16_t As[3 * 18 * 256];   // 27 KB
  int tid = threadIdx.x;
  int wave = tid >> 5, lane = tid & 31;
  int mlo = lane & 15, ch = lane >> 4;
  int tiles_x = Wc >> 4;
  int pt = blockIdx.x;
  int xt = pt % tiles_x;
  int tt = pt / tiles_x;
  int y = tt % Hc, b = tt / Hc;
  int x0 = xt * 16;
  // cooperative zero-padded stage: 3*18*256 = 13824 bf16 = 1728 v8bf chunks, 512 threads
  for (int cki = tid; cki < 1728; cki += 512) {
    int e = cki * 8;
    int r = e / (18 * 256);
    int rem = e % (18 * 256);
    int cidx = rem >> 8, c8 = rem & 255;
    int iy = y + r - 1;
    int ix = x0 + cidx - 1;
    v8bf v = zero8bf();
    if (iy >= 0 && iy < Hc && ix >= 0 && ix < Wc)
      v = *reinterpret_cast<const v8bf*>(in + ((size_t)(b * Hc + iy) * Wc + ix) * D_ + c8);
    *reinterpret_cast<v8bf*>(&As[e]) = v;
  }
  __syncthreads();
  int nt = wave;   // 16 waves = 16 n-tiles = full 256 outputs
  v8f acc = {};
  #pragma unroll
  for (int tap = 0; tap < 9; ++tap) {
    int dy = tap / 3, dx = tap % 3;   // 0..2 (halo-relative)
    const bf16_t* arow = &As[((dy * 18) + (mlo + dx)) * 256 + ch * 8];
    const bf16_t* wrow = wt + ((size_t)tap * 256 + nt * 16 + mlo) * 256 + ch * 8;
    #pragma unroll
    for (int k0 = 0; k0 < 256; k0 += 32) {
      v16bf a = load_a16(arow + k0);
      v16bf bb = load_b16(wrow + k0);
      acc = __builtin_amdgcn_wmma_f32_16x16x32_bf16(false, a, false, bb, (short)0, acc, false, false);
    }
  }
  float bs = bias[nt * 16 + mlo];
  int rbase = ch * 8;
  #pragma unroll
  for (int r = 0; r < 8; ++r)
    out[((size_t)(b * Hc + y) * Wc + (x0 + rbase + r)) * D_ + nt * 16 + mlo] = acc[r] + bs;
}

// ---------------- fused star-MLP: out += star(relu(X@fc1^T)) @ fc2^T ----------------
// X tile (8 KB) and H tile (32 KB) both live in LDS.
__global__ void fused_mlp_kernel(const bf16_t* __restrict__ X, const bf16_t* __restrict__ Wfc1,
                                 const bf16_t* __restrict__ Wfc2, const float* __restrict__ star_s,
                                 const float* __restrict__ star_b, float* __restrict__ out, int ntok) {
  __shared__ alignas(32) bf16_t Hs[16 * HID_];   // 32 KB
  __shared__ alignas(16) bf16_t Xs[16 * D_];     // 8 KB
  int tid = threadIdx.x;
  int wave = tid >> 5, lane = tid & 31;
  int mlo = lane & 15, ch = lane >> 4, rbase = ch * 8;
  int mt = blockIdx.x;
  if (mt * 16 >= ntok) return;
  float ss = star_s[0], sb = star_b[0];
  // stage X tile: 4096 bf16 = 512 v8bf chunks, 256 threads -> 2 each
  #pragma unroll
  for (int e = tid * 8; e < 16 * D_; e += 256 * 8) {
    int row = e >> 8, col = e & 255;
    *reinterpret_cast<v8bf*>(&Xs[e]) =
        *reinterpret_cast<const v8bf*>(X + (size_t)(mt * 16 + row) * D_ + col);
  }
  __syncthreads();
  const bf16_t* arow = &Xs[mlo * D_ + ch * 8];
  // phase 1: H[16,1024] = star(relu(X @ fc1^T)) -> LDS (each wave: 8 hidden tiles)
  for (int j = 0; j < 8; ++j) {
    int nt = wave * 8 + j;
    const bf16_t* brow = Wfc1 + (size_t)(nt * 16 + mlo) * D_ + ch * 8;
    v8f acc = {};
    #pragma unroll
    for (int k0 = 0; k0 < D_; k0 += 32) {
      v16bf a = load_a16(arow + k0);
      v16bf b = load_b16(brow + k0);
      acc = __builtin_amdgcn_wmma_f32_16x16x32_bf16(false, a, false, b, (short)0, acc, false, false);
    }
    #pragma unroll
    for (int r = 0; r < 8; ++r) {
      float h = fmaxf(acc[r], 0.f);
      h = ss * h * h + sb;
      Hs[(rbase + r) * HID_ + nt * 16 + mlo] = f2bf(h);
    }
  }
  __syncthreads();
  // phase 2: Y[16,256] = H @ fc2^T, accumulate into out (each wave: 2 n-tiles)
  for (int j = 0; j < 2; ++j) {
    int nt = wave + j * 8;
    const bf16_t* hrow = &Hs[mlo * HID_ + ch * 8];
    const bf16_t* brow = Wfc2 + (size_t)(nt * 16 + mlo) * HID_ + ch * 8;
    v8f acc = {};
    for (int k0 = 0; k0 < HID_; k0 += 32) {
      v16bf a = load_a16(hrow + k0);
      v16bf b = load_b16(brow + k0);
      acc = __builtin_amdgcn_wmma_f32_16x16x32_bf16(false, a, false, b, (short)0, acc, false, false);
    }
    #pragma unroll
    for (int r = 0; r < 8; ++r) {
      size_t oi = (size_t)(mt * 16 + rbase + r) * D_ + nt * 16 + mlo;
      out[oi] += acc[r];
    }
  }
}

// ---------------- host orchestration ----------------
extern "C" void kernel_launch(void* const* d_in, const int* in_sizes, int n_in,
                              void* d_out, int out_size, void* d_ws, size_t ws_size,
                              hipStream_t stream) {
  (void)in_sizes; (void)n_in; (void)out_size; (void)ws_size;
  const float* src   = (const float*)d_in[0];
  const float* pos   = (const float*)d_in[1];
  const float* fpn   = (const float*)d_in[2];
  const float* ref   = (const float*)d_in[3];
  const float* n1w   = (const float*)d_in[5];
  const float* n2w   = (const float*)d_in[6];
  const float* Wv    = (const float*)d_in[7];
  const float* bv    = (const float*)d_in[8];
  const float* Woff  = (const float*)d_in[9];
  const float* boff  = (const float*)d_in[10];
  const float* Wattn = (const float*)d_in[11];
  const float* battn = (const float*)d_in[12];
  const float* Wout  = (const float*)d_in[13];
  const float* bout  = (const float*)d_in[14];
  const float* convw = (const float*)d_in[15];
  const float* convb = (const float*)d_in[16];
  const float* fc1   = (const float*)d_in[17];
  const float* fc2   = (const float*)d_in[18];
  const float* stars = (const float*)d_in[19];
  const float* starb = (const float*)d_in[20];

  float* out_all = (float*)d_out;
  const int NT_SRC = B_ * LQ_;       // 10768
  const int NT_FPN = B_ * FH_ * FW_; // 32768
  const int NV     = B_ * SUMHW_;    // 10752

  char* p = (char*)d_ws;
  auto alloc = [&](size_t bytes) -> void* {
    void* r = (void*)p;
    p += (bytes + 255) & ~(size_t)255;
    return r;
  };
  bf16_t* query_bf   = (bf16_t*)alloc((size_t)NT_SRC * D_ * 2);
  bf16_t* valin_bf   = (bf16_t*)alloc((size_t)NV * D_ * 2);
  bf16_t* fpn2_bf    = (bf16_t*)alloc((size_t)NT_FPN * D_ * 2);
  bf16_t* wv_bf      = (bf16_t*)alloc((size_t)65536 * 2);
  bf16_t* woff_bf    = (bf16_t*)alloc((size_t)49152 * 2);
  bf16_t* wattn_bf   = (bf16_t*)alloc((size_t)24576 * 2);
  bf16_t* wout_bf    = (bf16_t*)alloc((size_t)65536 * 2);
  bf16_t* fc1_bf     = (bf16_t*)alloc((size_t)HID_ * D_ * 2);
  bf16_t* fc2_bf     = (bf16_t*)alloc((size_t)D_ * HID_ * 2);
  bf16_t* convwt_bf  = (bf16_t*)alloc((size_t)27 * 65536 * 2);
  float*  value_f    = (float*)alloc((size_t)NV * D_ * 4);
  float*  off_f      = (float*)alloc((size_t)NT_SRC * 192 * 4);
  float*  logit_f    = (float*)alloc((size_t)NT_SRC * 96 * 4);
  float*  attnout_f  = (float*)alloc((size_t)NT_SRC * D_ * 4);
  bf16_t* attnout_bf = (bf16_t*)alloc((size_t)NT_SRC * D_ * 2);
  float*  src2_f     = (float*)alloc((size_t)NT_SRC * D_ * 4);
  bf16_t* f32in_bf   = (bf16_t*)alloc((size_t)B_ * 1024 * D_ * 2);
  float*  out32_f    = (float*)alloc((size_t)B_ * 1024 * D_ * 4);
  bf16_t* f64in_bf   = (bf16_t*)alloc((size_t)B_ * 4096 * D_ * 2);
  float*  out64_f    = (float*)alloc((size_t)B_ * 4096 * D_ * 4);
  bf16_t* f128in_bf  = (bf16_t*)alloc((size_t)B_ * 16384 * D_ * 2);
  float*  out128_f   = (float*)alloc((size_t)B_ * 16384 * D_ * 4);
  bf16_t* xcat_bf    = (bf16_t*)alloc((size_t)(NT_SRC + NT_FPN) * D_ * 2);

  // weight casts
  cast_f2bf_kernel<<<256, 256, 0, stream>>>(Wv, wv_bf, 65536);
  cast_f2bf_kernel<<<192, 256, 0, stream>>>(Woff, woff_bf, 49152);
  cast_f2bf_kernel<<<96, 256, 0, stream>>>(Wattn, wattn_bf, 24576);
  cast_f2bf_kernel<<<256, 256, 0, stream>>>(Wout, wout_bf, 65536);
  cast_f2bf_kernel<<<1024, 256, 0, stream>>>(fc1, fc1_bf, HID_ * D_);
  cast_f2bf_kernel<<<1024, 256, 0, stream>>>(fc2, fc2_bf, D_ * HID_);
  convw_transpose_kernel<<<6912, 256, 0, stream>>>(convw, convwt_bf, 27 * 65536);

  // layernorms
  ln_src_kernel<<<NT_SRC / 4, 128, 0, stream>>>(src, pos, n1w, query_bf, valin_bf);
  ln_bf16_kernel<<<NT_FPN / 4, 128, 0, stream>>>(fpn, n1w, fpn2_bf, NT_FPN);

  // projections (WMMA GEMMs, K=256)
  gemm_bf16_kernel<<<dim3(NV / 16, 4), 128, 0, stream>>>(valin_bf, wv_bf, bv, value_f, NV, 256);
  gemm_bf16_kernel<<<dim3(NT_SRC / 16, 3), 128, 0, stream>>>(query_bf, woff_bf, boff, off_f, NT_SRC, 192);
  gemm_bf16_kernel<<<dim3(NT_SRC / 16, 2), 128, 0, stream>>>(query_bf, wattn_bf, battn, logit_f, NT_SRC, 96);

  // deformable sampling + output projection
  deform_sample_kernel<<<(B_ * LQ_ * NH_) / 4, 128, 0, stream>>>(value_f, off_f, logit_f, ref,
                                                                 attnout_f, attnout_bf);
  gemm_bf16_kernel<<<dim3(NT_SRC / 16, 4), 128, 0, stream>>>(attnout_bf, wout_bf, bout, src2_f,
                                                             NT_SRC, 256);

  // src residual + LN2 (writes d_out part 1 and first half of MLP input)
  residual_ln_kernel<<<NT_SRC / 4, 128, 0, stream>>>(src, src2_f, n2w, out_all, xcat_bf, NT_SRC);

  // FPN fusion: 16x16 feat -> 32 -> 64 -> 128 (feats are rows of src2_f / fpn2)
  upsample_add_kernel<<<(B_ * 32 * 32 * D_) / 256, 256, 0, stream>>>(
      src2_f, nullptr, (long)LQ_ * D_, R_ + 4096, src2_f, (long)LQ_ * D_, R_ + 5120,
      16, 16, 32, 32, f32in_bf);
  conv3x3_kernel<<<B_ * 32 * 2, 512, 0, stream>>>(f32in_bf, convwt_bf, convb, out32_f, 32, 32);

  upsample_add_kernel<<<(B_ * 64 * 64 * D_) / 256, 256, 0, stream>>>(
      src2_f, nullptr, (long)LQ_ * D_, R_ + 0, out32_f, (long)32 * 32 * D_, 0,
      32, 32, 64, 64, f64in_bf);
  conv3x3_kernel<<<B_ * 64 * 4, 512, 0, stream>>>(f64in_bf, convwt_bf + (size_t)9 * 65536,
                                                  convb + 256, out64_f, 64, 64);

  upsample_add_kernel<<<(B_ * 128 * 128 * D_) / 256, 256, 0, stream>>>(
      nullptr, fpn2_bf, (long)16384 * D_, 0, out64_f, (long)64 * 64 * D_, 0,
      64, 64, 128, 128, f128in_bf);
  conv3x3_kernel<<<B_ * 128 * 8, 512, 0, stream>>>(f128in_bf, convwt_bf + (size_t)18 * 65536,
                                                   convb + 512, out128_f, 128, 128);

  // fpn residual + LN2 (writes d_out part 2 and second half of MLP input)
  residual_ln_kernel<<<NT_FPN / 4, 128, 0, stream>>>(fpn, out128_f, n2w,
                                                     out_all + (size_t)NT_SRC * D_,
                                                     xcat_bf + (size_t)NT_SRC * D_, NT_FPN);

  // fused star-MLP over all 43536 tokens, accumulating into d_out
  fused_mlp_kernel<<<(NT_SRC + NT_FPN) / 16, 256, 0, stream>>>(xcat_bf, fc1_bf, fc2_bf,
                                                               stars, starb, out_all,
                                                               NT_SRC + NT_FPN);
}
